// SpaceTempGoG_detr_dad_50251117363233
// MI455X (gfx1250) — matrix operand order repaired and checked by reading, not compiled, not run on hardware
//
#include <hip/hip_runtime.h>
#include <hip/hip_bf16.h>
#include <math.h>

// ============================================================================
// SpaceTempGoG forward for MI455X (gfx1250, wave32).
// Dense matmuls -> v_wmma_f32_16x16x32_f16. Weights (all GEMM B operands) are
// pre-packed once per launch into WMMA fragment-ordered f16 in workspace, so
// the GEMM hot loop is: 2 LDS A-fragment loads + 2 coalesced global B-fragment
// loads (L2-resident) + 4 WMMA. A-tile staging is branchless (row-clamped,
// float4 vector loads on the K-interior path).
// ============================================================================

typedef __attribute__((ext_vector_type(16))) _Float16 v16h;
typedef __attribute__((ext_vector_type(8)))  float    v8f;

#define HC_  256
#define HEADS 4

// ------------------------------ device helpers ------------------------------
__device__ __forceinline__ float blk_red(float v, float* red, int domax) {
  // blockDim.x must be 256
  int t = threadIdx.x;
  red[t] = v; __syncthreads();
  #pragma unroll
  for (int o = 128; o > 0; o >>= 1) {
    if (t < o) red[t] = domax ? fmaxf(red[t], red[t + o]) : (red[t] + red[t + o]);
    __syncthreads();
  }
  float r = red[0]; __syncthreads();
  return r;
}

__device__ __forceinline__ void atomicMaxF(float* addr, float val) {
  unsigned int* ua = (unsigned int*)addr;
  unsigned int old = __float_as_uint(*addr);
  while (__uint_as_float(old) < val) {
    unsigned int prev = atomicCAS(ua, old, __float_as_uint(val));
    if (prev == old) break;
    old = prev;
  }
}

// --------------------------- weight pre-packing -----------------------------
// Pack f32 weight B[K,N] (row-major) into WMMA B-fragment order, f16:
//   idx = (((nt * nks) + ks) * 32 + lane) * 16 + e
//   col = nt*16 + (lane&15);  k = ks*32 + (lane>>4)*16 + e   (zero padded)
__global__ void pack_b_kernel(const float* __restrict__ B, int K, int N,
                              int nks, long total, _Float16* __restrict__ out) {
  long i = (long)blockIdx.x * blockDim.x + threadIdx.x;
  if (i >= total) return;
  int e    = (int)(i & 15);
  int lane = (int)((i >> 4) & 31);
  long r   = i >> 9;
  int ks   = (int)(r % nks);
  int nt   = (int)(r / nks);
  int col  = nt * 16 + (lane & 15);
  int k    = ks * 32 + ((lane >> 4) * 16) + e;
  out[i] = (_Float16)((k < K && col < N) ? B[(size_t)k * N + col] : 0.0f);
}

// ------------------------------ WMMA GEMM -----------------------------------
// C[M,N] = act( A[M,K] @ B[K,N] + bias + resid ),  B pre-packed (f16 fragments)
// Block: 128 thr = 4 waves; block tile 64x64; K-step 32; wave tile 32x32.
//   A (16x32 f16): lane l: row=l&15, K(e)=((l>>4)*16) + e
//   B (32x16 f16): lane l: col=l&15, K(e)=((l>>4)*16) + e   (packed layout)
//   D (16x16 f32): lane l: col=l&15, row(vi)=((l>>4)*8)+vi
__global__ __launch_bounds__(128)
void gemm_wmma_kernel(const float* __restrict__ A, int lda,
                      const _Float16* __restrict__ Bp, int nks,
                      float* __restrict__ C, int ldc,
                      int M, int N, int K,
                      const float* __restrict__ bias,
                      const float* __restrict__ resid, int ldr,
                      int act) {
  __shared__ __align__(32) _Float16 As[4][32][16];  // [m-tile][lane][elem]

  const int tid  = threadIdx.x;
  const int lane = tid & 31;
  const int wave = tid >> 5;        // 0..3
  const int wr   = wave >> 1;       // 0..1  (32-row half)
  const int wc   = wave & 1;        // 0..1  (32-col half)
  const int l15  = lane & 15;
  const int kbA  = (lane >> 4) * 16;
  const int rowBase = blockIdx.y * 64;
  const int colBase = blockIdx.x * 64;

  // Staging row (clamped: OOB rows stage valid garbage, masked in epilogue).
  const int growA = rowBase + wave * 16 + l15;
  const float* arow = A + (size_t)(growA < M ? growA : (M - 1)) * lda;

  // Packed B fragment base pointers for this wave's two n-tiles.
  const _Float16* bp0 = Bp + ((size_t)((colBase >> 4) + 2 * wc + 0) * nks * 32 + lane) * 16;
  const _Float16* bp1 = Bp + ((size_t)((colBase >> 4) + 2 * wc + 1) * nks * 32 + lane) * 16;

  v8f zf = {0.f, 0.f, 0.f, 0.f, 0.f, 0.f, 0.f, 0.f};
  v8f acc00 = zf, acc01 = zf, acc10 = zf, acc11 = zf;

  for (int ks = 0; ks < nks; ++ks) {
    const int k0 = ks * 32;
    v16h h;
    if (k0 + 32 <= K) {
      // interior: two 32B runs -> four float4 loads, branchless
      const float4* p0 = (const float4*)(arow + k0 + kbA);
      const float4* p1 = (const float4*)(arow + k0 + kbA + 8);
      float4 r0 = p0[0], r1 = p0[1], r2 = p1[0], r3 = p1[1];
      h[0] = (_Float16)r0.x;  h[1] = (_Float16)r0.y;
      h[2] = (_Float16)r0.z;  h[3] = (_Float16)r0.w;
      h[4] = (_Float16)r1.x;  h[5] = (_Float16)r1.y;
      h[6] = (_Float16)r1.z;  h[7] = (_Float16)r1.w;
      h[8]  = (_Float16)r2.x; h[9]  = (_Float16)r2.y;
      h[10] = (_Float16)r2.z; h[11] = (_Float16)r2.w;
      h[12] = (_Float16)r3.x; h[13] = (_Float16)r3.y;
      h[14] = (_Float16)r3.z; h[15] = (_Float16)r3.w;
    } else {
      // K tail: clamped loads + select-zero (no exec divergence)
      #pragma unroll
      for (int e = 0; e < 16; ++e) {
        int kk = k0 + kbA + e;
        float v = arow[kk < K ? kk : 0];
        h[e] = (_Float16)((kk < K) ? v : 0.0f);
      }
    }
    *(v16h*)&As[wave][lane][0] = h;
    __syncthreads();

    v16h a0 = *(const v16h*)&As[2 * wr + 0][lane][0];
    v16h a1 = *(const v16h*)&As[2 * wr + 1][lane][0];
    v16h b0 = *(const v16h*)(bp0 + (size_t)ks * 512);
    v16h b1 = *(const v16h*)(bp1 + (size_t)ks * 512);

    acc00 = __builtin_amdgcn_wmma_f32_16x16x32_f16(false, a0, false, b0, (short)0, acc00, false, false);
    acc01 = __builtin_amdgcn_wmma_f32_16x16x32_f16(false, a0, false, b1, (short)0, acc01, false, false);
    acc10 = __builtin_amdgcn_wmma_f32_16x16x32_f16(false, a1, false, b0, (short)0, acc10, false, false);
    acc11 = __builtin_amdgcn_wmma_f32_16x16x32_f16(false, a1, false, b1, (short)0, acc11, false, false);
    __syncthreads();
  }

  // ---- epilogue ----
  v8f accs[2][2] = {{acc00, acc01}, {acc10, acc11}};
  #pragma unroll
  for (int i = 0; i < 2; ++i) {
    #pragma unroll
    for (int j = 0; j < 2; ++j) {
      int col = colBase + wc * 32 + j * 16 + l15;
      if (col >= N) continue;
      #pragma unroll
      for (int vi = 0; vi < 8; ++vi) {
        int row = rowBase + wr * 32 + i * 16 + (lane >> 4) * 8 + vi;
        if (row >= M) continue;
        float v = accs[i][j][vi];
        if (bias)  v += bias[col];
        if (resid) v += resid[(size_t)row * ldr + col];
        if (act == 1)      v = fmaxf(v, 0.0f);
        else if (act == 2) v = (v > 0.0f) ? v : 0.2f * v;
        C[(size_t)row * ldc + col] = v;
      }
    }
  }
}

// ------------------------------ utility kernels -----------------------------
__global__ void fill_kernel(float* p, long n, float v) {
  long i = (long)blockIdx.x * blockDim.x + threadIdx.x;
  if (i < n) p[i] = v;
}

__global__ void copy_mat_kernel(const float* __restrict__ S, int lds_,
                                float* __restrict__ D, int ldd,
                                int rows, int cols) {
  long i = (long)blockIdx.x * blockDim.x + threadIdx.x;
  if (i >= (long)rows * cols) return;
  int r = (int)(i / cols), c = (int)(i % cols);
  D[(size_t)r * ldd + c] = S[(size_t)r * lds_ + c];
}

// per-column mean / rstd (biased var) over `rows`; one block per column
__global__ __launch_bounds__(256)
void col_stats_kernel(const float* __restrict__ X, int ld, int rows,
                      float* mean, float* rstd) {
  __shared__ float red[256];
  int c = blockIdx.x, t = threadIdx.x;
  float s = 0.f, s2 = 0.f;
  for (int r = t; r < rows; r += 256) {
    float v = X[(size_t)r * ld + c];
    s += v; s2 += v * v;
  }
  float S  = blk_red(s,  red, 0);
  float S2 = blk_red(s2, red, 0);
  if (t == 0) {
    float m = S / rows;
    float var = S2 / rows - m * m;
    mean[c] = m;
    rstd[c] = rsqrtf(var + 1e-5f);
  }
}

// in-place (x - mean)*rstd[*g+b] + optional activation, submatrix via ld
__global__ void norm_act_kernel(float* X, int ld, int rows, int cols,
                                const float* __restrict__ mean,
                                const float* __restrict__ rstd,
                                const float* __restrict__ g,
                                const float* __restrict__ b, int act) {
  long i = (long)blockIdx.x * blockDim.x + threadIdx.x;
  if (i >= (long)rows * cols) return;
  int r = (int)(i / cols), c = (int)(i % cols);
  float v = (X[(size_t)r * ld + c] - mean[c]) * rstd[c];
  if (g) v = v * g[c] + b[c];
  if (act == 1)      v = fmaxf(v, 0.0f);
  else if (act == 2) v = (v > 0.0f) ? v : 0.2f * v;
  X[(size_t)r * ld + c] = v;
}

// ------------------------- TransformerConv kernels --------------------------
__global__ void edge_logit_kernel(const float* __restrict__ q,
                                  const float* __restrict__ k,
                                  const int* __restrict__ src,
                                  const int* __restrict__ dst, int E,
                                  const float* __restrict__ ew, int ews, int ewo,
                                  const float* __restrict__ We,
                                  float* __restrict__ logit) {
  long i = (long)blockIdx.x * blockDim.x + threadIdx.x;
  if (i >= (long)E * HEADS) return;
  int e = (int)(i >> 2), h = (int)(i & 3);
  int sN = src[e], dN = dst[e];
  float w = ew ? ew[(size_t)e * ews + ewo] : 0.f;
  const float* qp = q + (size_t)dN * HC_ + h * 64;
  const float* kp = k + (size_t)sN * HC_ + h * 64;
  const float* wp = We ? (We + h * 64) : nullptr;
  float acc = 0.f;
  for (int c = 0; c < 64; ++c) {
    float kv = kp[c] + (wp ? w * wp[c] : 0.f);
    acc += qp[c] * kv;
  }
  logit[i] = acc * 0.125f;  // 1/sqrt(64)
}

__global__ void seg_max_kernel(const float* __restrict__ logit,
                               const int* __restrict__ dst, int E,
                               float* __restrict__ m) {
  long i = (long)blockIdx.x * blockDim.x + threadIdx.x;
  if (i >= (long)E * HEADS) return;
  int e = (int)(i >> 2), h = (int)(i & 3);
  atomicMaxF(&m[(size_t)dst[e] * HEADS + h], logit[i]);
}

__global__ void seg_expsum_kernel(float* __restrict__ logit,  // in-place -> alpha
                                  const int* __restrict__ dst, int E,
                                  const float* __restrict__ m,
                                  float* __restrict__ s) {
  long i = (long)blockIdx.x * blockDim.x + threadIdx.x;
  if (i >= (long)E * HEADS) return;
  int e = (int)(i >> 2), h = (int)(i & 3);
  float a = expf(logit[i] - m[(size_t)dst[e] * HEADS + h]);
  logit[i] = a;
  atomicAdd(&s[(size_t)dst[e] * HEADS + h], a);
}

__global__ void scatter_kernel(const float* __restrict__ v,
                               const int* __restrict__ src,
                               const int* __restrict__ dst, int E,
                               const float* __restrict__ alpha,
                               const float* __restrict__ s,
                               const float* __restrict__ ew, int ews, int ewo,
                               const float* __restrict__ We,
                               float* __restrict__ out) {
  long i = (long)blockIdx.x * blockDim.x + threadIdx.x;
  if (i >= (long)E * HC_) return;
  int e = (int)(i >> 8), ch = (int)(i & 255);
  int h = ch >> 6;
  int sN = src[e], dN = dst[e];
  float a = alpha[(size_t)e * HEADS + h] / (s[(size_t)dN * HEADS + h] + 1e-16f);
  float w = ew ? ew[(size_t)e * ews + ewo] : 0.f;
  float val = v[(size_t)sN * HC_ + ch] + (We ? w * We[ch] : 0.f);
  atomicAdd(&out[(size_t)dN * HC_ + ch], val * a);
}

// beta gate:  g = sigmoid([out,xr,out-xr] @ Wbeta);  res = g*xr + (1-g)*out
__global__ __launch_bounds__(256)
void beta_combine_kernel(const float* __restrict__ out,
                         const float* __restrict__ xr,
                         const float* __restrict__ Wb,
                         float* __restrict__ dstM, int lddst) {
  __shared__ float red[256];
  int n = blockIdx.x, t = threadIdx.x;
  float o = out[(size_t)n * HC_ + t];
  float x = xr[(size_t)n * HC_ + t];
  float part = o * Wb[t] + x * Wb[256 + t] + (o - x) * Wb[512 + t];
  float gsum = blk_red(part, red, 0);
  float g = 1.0f / (1.0f + expf(-gsum));
  dstM[(size_t)n * lddst + t] = g * x + (1.0f - g) * o;
}

__global__ void add_combine_kernel(const float* __restrict__ out,
                                   const float* __restrict__ xr,
                                   float* __restrict__ dstM, int lddst, int Nn) {
  long i = (long)blockIdx.x * blockDim.x + threadIdx.x;
  if (i >= (long)Nn * HC_) return;
  int n = (int)(i >> 8), c = (int)(i & 255);
  dstM[(size_t)n * lddst + c] = out[i] + xr[i];
}

// ------------------------- pooling kernels ----------------------------------
__global__ void seg_sum_kernel(const float* __restrict__ xin,
                               const int* __restrict__ src,
                               const int* __restrict__ dst, int E,
                               float* __restrict__ out) {
  long i = (long)blockIdx.x * blockDim.x + threadIdx.x;
  if (i >= (long)E * HC_) return;
  int e = (int)(i >> 8), ch = (int)(i & 255);
  atomicAdd(&out[(size_t)dst[e] * HC_ + ch], xin[(size_t)src[e] * HC_ + ch]);
}

__global__ __launch_bounds__(256)
void score_kernel(const float* __restrict__ aggr,
                  const float* __restrict__ ncross,
                  const float* __restrict__ prel,
                  const float* __restrict__ proot,
                  const float* __restrict__ pb,
                  float* __restrict__ score) {
  __shared__ float red[256];
  int n = blockIdx.x, t = threadIdx.x;
  float part = aggr[(size_t)n * HC_ + t] * prel[t] + ncross[(size_t)n * HC_ + t] * proot[t];
  float s = blk_red(part, red, 0);
  if (t == 0) score[n] = s + pb[0];
}

// SAGPool top-k (stable, ties -> lower index) + tanh gate + global max pool
__global__ __launch_bounds__(256)
void pool_topk_gmax_kernel(const float* __restrict__ score,
                           const float* __restrict__ ncross,
                           float* __restrict__ gembed, int nper, int kk) {
  __shared__ float sc[32];
  __shared__ int   sel[32];
  __shared__ float tv[32];
  int f = blockIdx.x, t = threadIdx.x;
  if (t < nper) sc[t] = score[f * nper + t];
  __syncthreads();
  if (t == 0) {
    bool used[32];
    for (int j = 0; j < nper; ++j) used[j] = false;
    for (int r = 0; r < kk; ++r) {
      int best = 0; float bv = -INFINITY;
      for (int j = 0; j < nper; ++j)
        if (!used[j] && sc[j] > bv) { bv = sc[j]; best = j; }
      used[best] = true; sel[r] = best; tv[r] = tanhf(bv);
    }
  }
  __syncthreads();
  float mx = -INFINITY;
  for (int r = 0; r < kk; ++r) {
    float val = ncross[(size_t)(f * nper + sel[r]) * HC_ + t] * tv[r];
    mx = fmaxf(mx, val);
  }
  gembed[(size_t)f * HC_ + t] = mx;
}

// ------------------------- encoder kernels ----------------------------------
// MHA over T=256 tokens, d=256, H=4 heads of 64. One block per (token, head).
__global__ __launch_bounds__(256)
void mha_attn_kernel(const float* __restrict__ qkv, float* __restrict__ out) {
  __shared__ float p[256];
  __shared__ float red[256];
  int tok = blockIdx.x >> 2, h = blockIdx.x & 3;
  int t = threadIdx.x;
  const float* qp = qkv + (size_t)tok * 768 + h * 64;
  const float* kp = qkv + (size_t)t * 768 + 256 + h * 64;
  float s = 0.f;
  for (int c = 0; c < 64; ++c) s += qp[c] * kp[c];
  s *= 0.125f;
  float mx = blk_red(s, red, 1);
  float ex = expf(s - mx);
  p[t] = ex;
  float sum = blk_red(ex, red, 0);   // trailing barrier -> p[] visible
  float inv = 1.0f / sum;
  int d = t & 63, qtr = t >> 6;
  float acc = 0.f;
  for (int j = qtr * 64; j < qtr * 64 + 64; ++j)
    acc += p[j] * qkv[(size_t)j * 768 + 512 + h * 64 + d];
  red[t] = acc; __syncthreads();
  if (t < 64)
    out[(size_t)tok * 256 + h * 64 + t] =
        (red[t] + red[t + 64] + red[t + 128] + red[t + 192]) * inv;
}

// row LayerNorm over 256 cols (biased var)
__global__ __launch_bounds__(256)
void ln_row_kernel(const float* __restrict__ X, const float* __restrict__ g,
                   const float* __restrict__ b, float* __restrict__ Y) {
  __shared__ float red[256];
  int r = blockIdx.x, t = threadIdx.x;
  float x = X[(size_t)r * 256 + t];
  float mean = blk_red(x, red, 0) * (1.0f / 256.0f);
  float dv = x - mean;
  float var = blk_red(dv * dv, red, 0) * (1.0f / 256.0f);
  Y[(size_t)r * 256 + t] = dv * rsqrtf(var + 1e-5f) * g[t] + b[t];
}

__global__ void final_out_kernel(const float* __restrict__ lg, float* __restrict__ out, int T) {
  int t = blockIdx.x * blockDim.x + threadIdx.x;
  if (t >= T) return;
  float a = lg[t * 2], b = lg[t * 2 + 1];
  out[t * 2] = a; out[t * 2 + 1] = b;
  float m = fmaxf(a, b);
  float ea = expf(a - m), eb = expf(b - m);
  float inv = 1.0f / (ea + eb);
  out[2 * T + t * 2] = ea * inv; out[2 * T + t * 2 + 1] = eb * inv;
}

// ------------------------------ param leaf map ------------------------------
// Top-level d_in order = setup_inputs() insertion order:
//   0:x 1:edge_index 2:img_feat 3:atten_feat 4:video_adj_list 5:edge_embeddings
//   6:temporal_adj_list 7:temporal_edge_w 8:batch_vec 9..151: params leaves
// params flattened jax-pytree style (dict keys sorted, lists in order):
enum {
  L_att_fc_b = 0, L_att_fc_w = 1, L_fc1_b = 2, L_fc1_w = 3, L_fc2_b = 4, L_fc2_w = 5,
  L_gc1s = 6,    // Wbeta We Wk Wq Wskip Wv bk bq bskip bv
  L_gc1t = 16,
  L_gc2att = 26, // Wk Wq Wskip Wv bk bq bskip bv
  L_gc2i3d = 34,
  L_gc2sg = 42,
  L_gccross = 50,
  L_gt = 58,     // 2 x 12 (W1 W2 Win Wout b1 b2 bin bout ln1_b ln1_g ln2_b ln2_g)
  L_img_fc_b = 82, L_img_fc_w = 83,
  L_ol_bn_b = 84, L_ol_bn_g = 85, L_ol_fc_b = 86, L_ol_fc_w = 87,
  L_pool_b = 88, L_pool_rel = 89, L_pool_root = 90,
  L_ta = 91, L_tt = 115,
  L_x_bn_b = 139, L_x_bn_g = 140, L_x_fc_b = 141, L_x_fc_w = 142,
  L_COUNT = 143
};

// ------------------------------ host helpers --------------------------------
static inline void gemm(hipStream_t st, const float* A, int lda,
                        const _Float16* Bp, float* C, int ldc,
                        int M, int N, int K,
                        const float* bias, const float* resid, int ldr, int act) {
  int nks = (K + 31) / 32;
  dim3 g((N + 63) / 64, (M + 63) / 64);
  gemm_wmma_kernel<<<g, dim3(128), 0, st>>>(A, lda, Bp, nks, C, ldc, M, N, K, bias, resid, ldr, act);
}

static inline void fillf(hipStream_t st, float* p, long n, float v) {
  fill_kernel<<<(unsigned)((n + 255) / 256), 256, 0, st>>>(p, n, v);
}

// per-call weight-pack cache (no static state; rebuilt identically every call)
struct PackCache {
  hipStream_t st;
  const float* PV[L_COUNT];
  _Float16* arena;
  size_t off;
  const _Float16* cached[L_COUNT];
  const _Float16* get(int leaf, int K, int N) {
    if (cached[leaf]) return cached[leaf];
    int nks = (K + 31) / 32;
    int nnt = 4 * ((N + 63) / 64);              // pad to block width (4 n-tiles)
    size_t n = (size_t)nnt * nks * 512;
    _Float16* p = arena + off;
    off += (n + 31) & ~(size_t)31;              // keep 32B alignment
    pack_b_kernel<<<(unsigned)((n + 255) / 256), 256, 0, st>>>(PV[leaf], K, N, nks, (long)n, p);
    cached[leaf] = p;
    return p;
  }
};

struct TP {
  const _Float16 *Wq, *Wk, *Wv, *Ws;           // packed weights
  const float *bq, *bk, *bv, *bs, *We, *Wb;    // raw f32 leaves
};

static TP tp_full(PackCache& pc, int base, int din) {  // with We + Wbeta
  TP t;
  t.Wb = pc.PV[base + 0]; t.We = pc.PV[base + 1];
  t.Wk = pc.get(base + 2, din, HC_); t.Wq = pc.get(base + 3, din, HC_);
  t.Ws = pc.get(base + 4, din, HC_); t.Wv = pc.get(base + 5, din, HC_);
  t.bk = pc.PV[base + 6]; t.bq = pc.PV[base + 7];
  t.bs = pc.PV[base + 8]; t.bv = pc.PV[base + 9];
  return t;
}
static TP tp_plain(PackCache& pc, int base, int din) {
  TP t;
  t.Wb = nullptr; t.We = nullptr;
  t.Wk = pc.get(base + 0, din, HC_); t.Wq = pc.get(base + 1, din, HC_);
  t.Ws = pc.get(base + 2, din, HC_); t.Wv = pc.get(base + 3, din, HC_);
  t.bk = pc.PV[base + 4]; t.bq = pc.PV[base + 5];
  t.bs = pc.PV[base + 6]; t.bv = pc.PV[base + 7];
  return t;
}

struct TWs {
  float *q, *k, *v, *xr, *tout, *logit, *m, *s;
};

static void tconv_run(hipStream_t st, const float* X, int ldx, int din, int Nn,
                      const int* esrc, const int* edst, int E,
                      const float* ew, int ews, int ewo,
                      const TP& p, float* dstM, int lddst, const TWs& w) {
  gemm(st, X, ldx, p.Wq, w.q,  HC_, Nn, HC_, din, p.bq, nullptr, 0, 0);
  gemm(st, X, ldx, p.Wk, w.k,  HC_, Nn, HC_, din, p.bk, nullptr, 0, 0);
  gemm(st, X, ldx, p.Wv, w.v,  HC_, Nn, HC_, din, p.bv, nullptr, 0, 0);
  gemm(st, X, ldx, p.Ws, w.xr, HC_, Nn, HC_, din, p.bs, nullptr, 0, 0);

  fillf(st, w.m, (long)Nn * HEADS, -INFINITY);
  fillf(st, w.s, (long)Nn * HEADS, 0.0f);
  fillf(st, w.tout, (long)Nn * HC_, 0.0f);

  long eh = (long)E * HEADS;
  edge_logit_kernel<<<(unsigned)((eh + 255) / 256), 256, 0, st>>>(w.q, w.k, esrc, edst, E, ew, ews, ewo, p.We, w.logit);
  seg_max_kernel<<<(unsigned)((eh + 255) / 256), 256, 0, st>>>(w.logit, edst, E, w.m);
  seg_expsum_kernel<<<(unsigned)((eh + 255) / 256), 256, 0, st>>>(w.logit, edst, E, w.m, w.s);
  long ec = (long)E * HC_;
  scatter_kernel<<<(unsigned)((ec + 255) / 256), 256, 0, st>>>(w.v, esrc, edst, E, w.logit, w.s, ew, ews, ewo, p.We, w.tout);

  if (p.Wb) {
    beta_combine_kernel<<<Nn, 256, 0, st>>>(w.tout, w.xr, p.Wb, dstM, lddst);
  } else {
    long nc = (long)Nn * HC_;
    add_combine_kernel<<<(unsigned)((nc + 255) / 256), 256, 0, st>>>(w.tout, w.xr, dstM, lddst, Nn);
  }
}

static void inorm_lrelu(hipStream_t st, float* X, int ld, int rows, int cols,
                        float* cmean, float* crstd) {
  col_stats_kernel<<<cols, 256, 0, st>>>(X, ld, rows, cmean, crstd);
  long n = (long)rows * cols;
  norm_act_kernel<<<(unsigned)((n + 255) / 256), 256, 0, st>>>(X, ld, rows, cols, cmean, crstd, nullptr, nullptr, 2);
}

// one post-LN transformer encoder layer (d=256, T tokens)
static void enc_layer(hipStream_t st, const float* const* L,
                      const _Float16* pWin, const _Float16* pWout,
                      const _Float16* pW1, const _Float16* pW2,
                      float* tio, int T,
                      float* tB, float* qkv, float* attn, float* ybuf, float* ffn) {
  // leaf order (sorted): W1=0 W2=1 Win=2 Wout=3 b1=4 b2=5 bin=6 bout=7 ln1_b=8 ln1_g=9 ln2_b=10 ln2_g=11
  gemm(st, tio, 256, pWin, qkv, 768, T, 768, 256, L[6], nullptr, 0, 0);
  mha_attn_kernel<<<T * HEADS, 256, 0, st>>>(qkv, attn);
  gemm(st, attn, 256, pWout, ybuf, 256, T, 256, 256, L[7], tio, 256, 0);
  ln_row_kernel<<<T, 256, 0, st>>>(ybuf, L[9], L[8], tB);
  gemm(st, tB, 256, pW1, ffn, 2048, T, 2048, 256, L[4], nullptr, 0, 1);
  gemm(st, ffn, 2048, pW2, ybuf, 256, T, 256, 2048, L[5], tB, 256, 0);
  ln_row_kernel<<<T, 256, 0, st>>>(ybuf, L[11], L[10], tio);
}

// ------------------------------ entry point ---------------------------------
extern "C" void kernel_launch(void* const* d_in, const int* in_sizes, int n_in,
                              void* d_out, int out_size, void* d_ws, size_t ws_size,
                              hipStream_t stream) {
  const float* x    = (const float*)d_in[0];
  const int*   ei   = (const int*)d_in[1];
  const float* img  = (const float*)d_in[2];
  const float* att  = (const float*)d_in[3];
  const int*   vadj = (const int*)d_in[4];
  const float* eemb = (const float*)d_in[5];
  const int*   tadj = (const int*)d_in[6];
  const float* tew  = (const float*)d_in[7];
  // d_in[8] = batch_vec (unused: uniform NOBJ per frame)

  const int ES = in_sizes[1] / 2;     // 147456 spatial edges
  const int T  = in_sizes[2] / 2048;  // 256 frames
  const int ET = in_sizes[6] / 2;     // 12240 temporal edges
  const int EV = in_sizes[4] / 2;     // 510 video edges
  const int Nn = in_sizes[8];         // 6144 nodes
  const int nper = Nn / T;            // 24
  const int kk = (int)((0.8 * nper) + 0.999999);  // ceil(0.8*24)=20

  // ---------------- workspace layout (floats) ----------------
  float* ws = (float*)d_ws;
  size_t off = 0;
  auto alloc = [&](size_t n) { float* p = ws + off; off += (n + 15) & ~(size_t)15; return p; };
  float* xc     = alloc((size_t)Nn * 320);
  float* qb     = alloc((size_t)Nn * HC_);
  float* kb     = alloc((size_t)Nn * HC_);
  float* vb     = alloc((size_t)Nn * HC_);
  float* xrb    = alloc((size_t)Nn * HC_);
  float* toutb  = alloc((size_t)Nn * HC_);
  float* mbuf   = alloc((size_t)Nn * HEADS);
  float* sbuf   = alloc((size_t)Nn * HEADS);
  float* logit  = alloc((size_t)ES * HEADS);
  float* nsnt   = alloc((size_t)Nn * 512);
  float* ncross = alloc((size_t)Nn * HC_);
  float* aggrb  = alloc((size_t)Nn * HC_);
  float* score  = alloc((size_t)Nn);
  float* cmean  = alloc(512);
  float* crstd  = alloc(512);
  float* gemb   = alloc((size_t)T * HC_);
  float* tA     = alloc((size_t)T * HC_);
  float* tB     = alloc((size_t)T * HC_);
  float* qkv    = alloc((size_t)T * 768);
  float* attnb  = alloc((size_t)T * HC_);
  float* ybuf   = alloc((size_t)T * HC_);
  float* ffn    = alloc((size_t)T * 2048);
  float* feat   = alloc((size_t)T * 1024);
  float* h1     = alloc((size_t)T * 128);
  float* lgts   = alloc((size_t)T * 2);
  // f16 weight-pack arena (32B aligned; ~12M halves needed for this net)
  _Float16* packArena = (_Float16*)(ws + off);
  (void)ws_size; (void)n_in; (void)out_size;

  PackCache pc;
  pc.st = stream; pc.arena = packArena; pc.off = 0;
  for (int i = 0; i < L_COUNT; ++i) { pc.PV[i] = (const float*)d_in[9 + i]; pc.cached[i] = nullptr; }
  const float* const* PV = pc.PV;

  TWs tw = {qb, kb, vb, xrb, toutb, logit, mbuf, sbuf};

  // ---- input projections + BN + lrelu -> xc [Nn x 320] ----
  gemm(stream, x, 2348, pc.get(L_x_fc_w, 2048, 256), xc, 320, Nn, 256, 2048, PV[L_x_fc_b], nullptr, 0, 0);
  col_stats_kernel<<<256, 256, 0, stream>>>(xc, 320, Nn, cmean, crstd);
  norm_act_kernel<<<(unsigned)(((long)Nn * 256 + 255) / 256), 256, 0, stream>>>(
      xc, 320, Nn, 256, cmean, crstd, PV[L_x_bn_g], PV[L_x_bn_b], 2);

  gemm(stream, x + 2048, 2348, pc.get(L_ol_fc_w, 300, 64), xc + 256, 320, Nn, 64, 300, PV[L_ol_fc_b], nullptr, 0, 0);
  col_stats_kernel<<<64, 256, 0, stream>>>(xc + 256, 320, Nn, cmean, crstd);
  norm_act_kernel<<<(unsigned)(((long)Nn * 64 + 255) / 256), 256, 0, stream>>>(
      xc + 256, 320, Nn, 64, cmean, crstd, PV[L_ol_bn_g], PV[L_ol_bn_b], 2);

  // ---- gc1 spatial / temporal (edge attr + beta) ----
  TP p1s = tp_full(pc, L_gc1s, 320), p1t = tp_full(pc, L_gc1t, 320);
  tconv_run(stream, xc, 320, 320, Nn, ei, ei + ES, ES, eemb, 4, 3, p1s, nsnt, 512, tw);
  inorm_lrelu(stream, nsnt, 512, Nn, 256, cmean, crstd);
  tconv_run(stream, xc, 320, 320, Nn, tadj, tadj + ET, ET, tew, 1, 0, p1t, nsnt + 256, 512, tw);
  inorm_lrelu(stream, nsnt + 256, 512, Nn, 256, cmean, crstd);

  // ---- gc_cross on concat [n_s | n_t] ----
  TP pcr = tp_plain(pc, L_gccross, 512);
  tconv_run(stream, nsnt, 512, 512, Nn, ei, ei + ES, ES, nullptr, 0, 0, pcr, ncross, HC_, tw);
  inorm_lrelu(stream, ncross, HC_, Nn, 256, cmean, crstd);

  // ---- SAGPool + global max pool -> g_embed [T x 256] ----
  fillf(stream, aggrb, (long)Nn * HC_, 0.0f);
  seg_sum_kernel<<<(unsigned)(((long)ES * HC_ + 255) / 256), 256, 0, stream>>>(ncross, ei, ei + ES, ES, aggrb);
  score_kernel<<<Nn, 256, 0, stream>>>(aggrb, ncross, PV[L_pool_rel], PV[L_pool_root], PV[L_pool_b], score);
  pool_topk_gmax_kernel<<<T, 256, 0, stream>>>(score, ncross, gemb, nper, kk);

  // ---- g_trans = encoder(g_embed, gt) -> feat[:, 768:1024] ----
  copy_mat_kernel<<<(unsigned)(((long)T * 256 + 255) / 256), 256, 0, stream>>>(gemb, 256, tA, 256, T, 256);
  for (int l = 0; l < 2; ++l) {
    int b = L_gt + 12 * l;
    enc_layer(stream, &PV[b], pc.get(b + 2, 256, 768), pc.get(b + 3, 256, 256),
              pc.get(b + 0, 256, 2048), pc.get(b + 1, 2048, 256),
              tA, T, tB, qkv, attnb, ybuf, ffn);
  }
  copy_mat_kernel<<<(unsigned)(((long)T * 256 + 255) / 256), 256, 0, stream>>>(tA, 256, feat + 768, 1024, T, 256);

  // ---- img path: fc -> encoder(tt) -> gc2_i3d -> feat[:, 256:512] ----
  gemm(stream, img, 2048, pc.get(L_img_fc_w, 2048, 256), tA, 256, T, 256, 2048, PV[L_img_fc_b], nullptr, 0, 0);
  for (int l = 0; l < 2; ++l) {
    int b = L_tt + 12 * l;
    enc_layer(stream, &PV[b], pc.get(b + 2, 256, 768), pc.get(b + 3, 256, 256),
              pc.get(b + 0, 256, 2048), pc.get(b + 1, 2048, 256),
              tA, T, tB, qkv, attnb, ybuf, ffn);
  }
  TP pi3d = tp_plain(pc, L_gc2i3d, 256);
  tconv_run(stream, tA, 256, 256, T, vadj, vadj + EV, EV, nullptr, 0, 0, pi3d, feat + 256, 1024, tw);
  inorm_lrelu(stream, feat + 256, 1024, T, 256, cmean, crstd);

  // ---- attention path: fc -> encoder(ta) -> gc2_att -> feat[:, 512:768] ----
  gemm(stream, att, 2048, pc.get(L_att_fc_w, 2048, 256), tA, 256, T, 256, 2048, PV[L_att_fc_b], nullptr, 0, 0);
  for (int l = 0; l < 2; ++l) {
    int b = L_ta + 12 * l;
    enc_layer(stream, &PV[b], pc.get(b + 2, 256, 768), pc.get(b + 3, 256, 256),
              pc.get(b + 0, 256, 2048), pc.get(b + 1, 2048, 256),
              tA, T, tB, qkv, attnb, ybuf, ffn);
  }
  TP patt = tp_plain(pc, L_gc2att, 256);
  tconv_run(stream, tA, 256, 256, T, vadj, vadj + EV, EV, nullptr, 0, 0, patt, feat + 512, 1024, tw);
  inorm_lrelu(stream, feat + 512, 1024, T, 256, cmean, crstd);

  // ---- scene-graph path: gc2_sg on g_embed -> feat[:, 0:256] ----
  TP psg = tp_plain(pc, L_gc2sg, 256);
  tconv_run(stream, gemb, 256, 256, T, vadj, vadj + EV, EV, nullptr, 0, 0, psg, feat, 1024, tw);
  inorm_lrelu(stream, feat, 1024, T, 256, cmean, crstd);

  // ---- head: fc1 (lrelu) -> fc2 -> logits + softmax ----
  gemm(stream, feat, 1024, pc.get(L_fc1_w, 1024, 128), h1, 128, T, 128, 1024, PV[L_fc1_b], nullptr, 0, 2);
  gemm(stream, h1, 128, pc.get(L_fc2_w, 128, 2), lgts, 2, T, 2, 128, PV[L_fc2_b], nullptr, 0, 0);
  final_out_kernel<<<(T + 255) / 256, 256, 0, stream>>>(lgts, (float*)d_out, T);
}
